// SchNetAvg_15676630630708
// MI455X (gfx1250) — matrix operand order repaired and checked by reading, not compiled, not run on hardware
//
#include <hip/hip_runtime.h>
#include <hip/hip_bf16.h>

// ---------------------------------------------------------------------------
// SchNet forward for MI455X (gfx1250, wave32, WMMA).
//
// Input flattening assumption (setup_inputs() insertion order, params walked
// depth-first in insertion order):
//   0: z[N] i32            1: position[N*3] f32   2: edge_index[2*E] i32
//   3: batch[N] i32        4: solvent[G] i32
//   5: emb_z[100*64]       6: emb_solv[4*64]
//   7: solv_w1[64*64]  8: solv_b1[64]  9: solv_w2[64*32] 10: solv_b2[32]
//   11..20  inter[0]: lin1_w, lin1_b, g0_w[50*64], g0_b, g1_w[64*64], g1_b,
//                     mlp0_w, mlp0_b, mlp1_w, mlp1_b
//   21..30  inter[1]: same
//   31: post0_w 32: post0_b 33: post1_w 34: post1_b
//   35: p2_w1[96*128] 36: p2_b1[128] 37: p2_w2[128*32] 38: p2_b2[32]
//   39: p2_w3[32]     40: p2_b3[1]
// Output: [G] f32.
//
// Edge GEMMs are oriented as  D[outfeat, edge] = W^T(A) x act(B):
//   A operand = weights (M=16 out-features, K=32 chunk), f16
//   B operand = activations (K=32 chunk, N=16 edges), f16
// so each lane owns ONE edge and 8 consecutive features per C tile:
// src/dst stay in-lane (no bpermute), m[src] gathers become float4 loads,
// and the v[dst] atomics hit consecutive addresses.
// ---------------------------------------------------------------------------

typedef __attribute__((ext_vector_type(16))) _Float16 v16h;
typedef __attribute__((ext_vector_type(8)))  float    v8f;

#define FEAT   64
#define NRBF   50
#define GAMMA_ 10.0f
#define LOG2_  0.6931471805599453f
#define EDGE_WAVES 4
#define HPITCH 18   // LDS row pitch (floats): half-wave groups hit disjoint banks

// Shifted softplus, branch-free: ssp(x) = max(x,0) + log(1+exp(-|x|)) - log(2).
__device__ __forceinline__ float sspf(float v) {
    float t = __expf(-fabsf(v));          // in (0,1]
    return fmaxf(v, 0.0f) + __logf(1.0f + t) - LOG2_;
}

// ---------------------------------------------------------------------------
// Pack a row-major f32 weight W[K][64] (Kact rows, zero-padded to K=64) into
// a 16-bit WMMA A-operand layout with contiguous k (weights transposed: M = out-feature).
//   out index = ((f*2 + c)*32 + lane)*16 + h,  f = 16-feature tile, c = K chunk
//   M row  n = f*16 + (lane&15)
//   K      k = c*32 + hi16 + h,  hi16 = (lane&16)?16:0
// foldBias: put bias[n] into the padded row k==63 (driven by B input of 1.0).
// ---------------------------------------------------------------------------
__global__ void pack_a_kernel(const float* __restrict__ W, const float* __restrict__ bias,
                              _Float16* __restrict__ out, int Kact, int foldBias) {
    int i = blockIdx.x * blockDim.x + threadIdx.x;
    if (i >= 4096) return;
    int h    = i & 15;
    int lane = (i >> 4) & 31;
    int fc   = i >> 9;           // f*2 + c
    int c    = fc & 1;
    int f    = fc >> 1;
    int n    = f * 16 + (lane & 15);
    int hi8  = (lane & 16) ? 8 : 0;
    int k    = c * 32 + 2 * hi8 + h;
    float val = 0.0f;
    if (k < Kact)                 val = W[k * FEAT + n];
    else if (foldBias && k == 63) val = bias[n];
    out[i] = (_Float16)val;
}

__global__ void fill_kernel(float* __restrict__ p, int n, float v) {
    int i = blockIdx.x * blockDim.x + threadIdx.x;
    if (i < n) p[i] = v;
}

__global__ void embed_kernel(const int* __restrict__ z, const float* __restrict__ emb,
                             float* __restrict__ x, int n) {
    int i = blockIdx.x * blockDim.x + threadIdx.x;
    if (i >= n * FEAT) return;
    int node = i >> 6, f = i & 63;
    x[i] = emb[z[node] * FEAT + f];
}

// y = [res +] (ssp?)(x @ W + b);  W row-major [din][dout]; in-place residual ok.
__global__ void linear_kernel(const float* __restrict__ x, const float* __restrict__ W,
                              const float* __restrict__ b, float* y,
                              const float* res, int n, int din, int dout, int act) {
    int i = blockIdx.x * blockDim.x + threadIdx.x;
    if (i >= n * dout) return;
    int node = i / dout, f = i % dout;
    float acc = b[f];
    const float* xr = x + node * din;
    for (int k = 0; k < din; ++k) acc = fmaf(xr[k], W[k * dout + f], acc);
    if (act) acc = sspf(acc);
    if (res) acc += res[i];
    y[i] = acc;
}

// ---------------------------------------------------------------------------
// Fused edge kernel: one 16-edge tile per wave iteration.
//   rbf(B:64x16, bias slot k=63) --WMMA--> ssp --LDS restage--> --WMMA-->
//   ssp(+b1) = g[feat][edge];   v[dst][f] += m[src][f] * g   (f32 atomics)
// ---------------------------------------------------------------------------
__global__ void __launch_bounds__(EDGE_WAVES * 32)
edge_kernel(const int* __restrict__ ei, const float* __restrict__ pos,
            const float* __restrict__ m,
            const _Float16* __restrict__ W0p, const _Float16* __restrict__ W1p,
            const float* __restrict__ b1v,
            float* __restrict__ vout, int E_) {
    __shared__ float hbuf[EDGE_WAVES][64 * HPITCH];
    const int lane = threadIdx.x & 31;
    const int w    = threadIdx.x >> 5;
    const int ecol = lane & 15;               // this lane's edge within the tile
    const int hi8  = (lane & 16) ? 8 : 0;     // C/D row offset for upper half-wave
    const int khi  = (lane & 16) ? 16 : 0;    // B-operand K offset for upper half-wave

    // Preload both layers' packed A-operand weights (loop-invariant).
    v16h W0[4][2], W1[4][2];
#pragma unroll
    for (int f = 0; f < 4; ++f)
#pragma unroll
        for (int c = 0; c < 2; ++c) {
            W0[f][c] = *(const v16h*)(W0p + ((f * 2 + c) * 32 + lane) * 16);
            W1[f][c] = *(const v16h*)(W1p + ((f * 2 + c) * 32 + lane) * 16);
        }
    // Layer-1 bias for the 8 consecutive features this lane owns per tile.
    float bias1a[4][8];
#pragma unroll
    for (int f = 0; f < 4; ++f)
#pragma unroll
        for (int vv = 0; vv < 8; ++vv)
            bias1a[f][vv] = b1v[f * 16 + hi8 + vv];

    const int ntiles = (E_ + 15) >> 4;
    for (int tile = blockIdx.x * EDGE_WAVES + w; tile < ntiles;
         tile += gridDim.x * EDGE_WAVES) {
        int e      = tile * 16 + ecol;
        bool valid = (e < E_);
        int ec     = valid ? e : (E_ - 1);
        int sN     = ei[ec];
        int dN     = ei[E_ + ec];

        float dx = pos[dN * 3 + 0] - pos[sN * 3 + 0];
        float dy = pos[dN * 3 + 1] - pos[sN * 3 + 1];
        float dz = pos[dN * 3 + 2] - pos[sN * 3 + 2];
        float dist = sqrtf(dx * dx + dy * dy + dz * dz);

        // RBF into 16-bit B-operand layout: lane holds K = chunk*32 + khi + h
        // for its own edge column. Chunk0 K=0..31 (< NRBF always); chunk1 has
        // the zero pad and the bias-driver 1.0 at K=63.
        v16h Brbf0, Brbf1;
#pragma unroll
        for (int h = 0; h < 16; ++h) {
            int k0 = khi + h;                       // 0..31
            float u0 = dist - 0.1f * (float)k0;
            Brbf0[h] = (_Float16)__expf(-GAMMA_ * u0 * u0);
            int k1 = 32 + khi + h;                  // 32..63
            float u1 = dist - 0.1f * (float)k1;
            float r1 = (k1 < NRBF) ? __expf(-GAMMA_ * u1 * u1)
                                   : ((k1 == 63) ? 1.0f : 0.0f);
            Brbf1[h] = (_Float16)r1;
        }

        // Layer g0: D[outfeat, edge]; bias folded into K=63 row. SSP -> LDS.
#pragma unroll
        for (int f = 0; f < 4; ++f) {
            v8f acc = {};
            acc = __builtin_amdgcn_wmma_f32_16x16x32_f16(false, W0[f][0], false, Brbf0,
                                                         (short)0, acc, false, false);
            acc = __builtin_amdgcn_wmma_f32_16x16x32_f16(false, W0[f][1], false, Brbf1,
                                                         (short)0, acc, false, false);
#pragma unroll
            for (int vv = 0; vv < 8; ++vv)
                hbuf[w][(f * 16 + hi8 + vv) * HPITCH + ecol] = sspf(acc[vv]);
        }

        // Re-stage h[feat][edge] -> layer-1 B operand (same-wave LDS, in-order).
        v16h Bh0, Bh1;
#pragma unroll
        for (int h = 0; h < 16; ++h) {
            Bh0[h] = (_Float16)hbuf[w][(khi + h) * HPITCH + ecol];
            Bh1[h] = (_Float16)hbuf[w][(32 + khi + h) * HPITCH + ecol];
        }

        // Layer g1 + bias + SSP, then msg = m[src]*g -> coalesced atomic scatter.
#pragma unroll
        for (int f = 0; f < 4; ++f) {
            v8f acc = {};
            acc = __builtin_amdgcn_wmma_f32_16x16x32_f16(false, W1[f][0], false, Bh0,
                                                         (short)0, acc, false, false);
            acc = __builtin_amdgcn_wmma_f32_16x16x32_f16(false, W1[f][1], false, Bh1,
                                                         (short)0, acc, false, false);
            if (valid) {
                const float4* mp = (const float4*)(m + sN * FEAT + f * 16 + hi8);
                float4 m0 = mp[0];
                float4 m1 = mp[1];
                float* vp = vout + dN * FEAT + f * 16 + hi8;
                atomicAdd(vp + 0, m0.x * sspf(acc[0] + bias1a[f][0]));
                atomicAdd(vp + 1, m0.y * sspf(acc[1] + bias1a[f][1]));
                atomicAdd(vp + 2, m0.z * sspf(acc[2] + bias1a[f][2]));
                atomicAdd(vp + 3, m0.w * sspf(acc[3] + bias1a[f][3]));
                atomicAdd(vp + 4, m1.x * sspf(acc[4] + bias1a[f][4]));
                atomicAdd(vp + 5, m1.y * sspf(acc[5] + bias1a[f][5]));
                atomicAdd(vp + 6, m1.z * sspf(acc[6] + bias1a[f][6]));
                atomicAdd(vp + 7, m1.w * sspf(acc[7] + bias1a[f][7]));
            }
        }
    }
}

// scatter-sum of node features into per-graph accumulators + counts
__global__ void scatter_kernel(const float* __restrict__ xf, const int* __restrict__ batch,
                               float* __restrict__ s, float* __restrict__ cnt, int n) {
    int i = blockIdx.x * blockDim.x + threadIdx.x;
    if (i >= n * FEAT) return;
    int node = i >> 6, f = i & 63;
    int g = batch[node];
    atomicAdd(s + g * FEAT + f, xf[i]);
    if (f == 0) atomicAdd(cnt + g, 1.0f);
}

// solvent branch: emb_solv[solvent] -> lin64 -> ssp -> lin32   (one block/graph)
__global__ void solvent_kernel(const int* __restrict__ solvent, const float* __restrict__ emb,
                               const float* __restrict__ w1, const float* __restrict__ b1,
                               const float* __restrict__ w2, const float* __restrict__ b2,
                               float* __restrict__ solv) {
    __shared__ float e0[64], h1[64];
    int g = blockIdx.x, t = threadIdx.x;
    e0[t] = emb[solvent[g] * 64 + t];
    __syncthreads();
    float a = b1[t];
    for (int k = 0; k < 64; ++k) a = fmaf(e0[k], w1[k * 64 + t], a);
    h1[t] = sspf(a);
    __syncthreads();
    if (t < 32) {
        float a2 = b2[t];
        for (int k = 0; k < 64; ++k) a2 = fmaf(h1[k], w2[k * 32 + t], a2);
        solv[g * 32 + t] = a2;
    }
}

// head: concat(mean-pool, solv)[96] -> 128 -> ssp -> 32 -> ssp -> 1
__global__ void head_kernel(const float* __restrict__ s, const float* __restrict__ cnt,
                            const float* __restrict__ solv,
                            const float* __restrict__ w1, const float* __restrict__ b1,
                            const float* __restrict__ w2, const float* __restrict__ b2,
                            const float* __restrict__ w3, const float* __restrict__ b3,
                            float* __restrict__ out) {
    __shared__ float cc[96], h1[128], h2[32];
    int g = blockIdx.x, t = threadIdx.x;
    if (t < 64)       cc[t] = s[g * 64 + t] / fmaxf(cnt[g], 1.0f);
    else if (t < 96)  cc[t] = solv[g * 32 + (t - 64)];
    __syncthreads();
    float a = b1[t];
    for (int k = 0; k < 96; ++k) a = fmaf(cc[k], w1[k * 128 + t], a);
    h1[t] = sspf(a);
    __syncthreads();
    if (t < 32) {
        float a2 = b2[t];
        for (int k = 0; k < 128; ++k) a2 = fmaf(h1[k], w2[k * 32 + t], a2);
        h2[t] = sspf(a2);
    }
    __syncthreads();
    if (t == 0) {
        float a3 = b3[0];
        for (int k = 0; k < 32; ++k) a3 = fmaf(h2[k], w3[k], a3);
        out[g] = a3;
    }
}

// ---------------------------------------------------------------------------
extern "C" void kernel_launch(void* const* d_in, const int* in_sizes, int n_in,
                              void* d_out, int out_size, void* d_ws, size_t ws_size,
                              hipStream_t stream) {
    const int*   z       = (const int*)  d_in[0];
    const float* pos     = (const float*)d_in[1];
    const int*   ei      = (const int*)  d_in[2];
    const int*   batch   = (const int*)  d_in[3];
    const int*   solvent = (const int*)  d_in[4];
    const float* emb_z   = (const float*)d_in[5];
    const float* emb_s   = (const float*)d_in[6];
    const float* sw1 = (const float*)d_in[7];
    const float* sb1 = (const float*)d_in[8];
    const float* sw2 = (const float*)d_in[9];
    const float* sb2 = (const float*)d_in[10];

    const int N = in_sizes[0];
    const int E = in_sizes[2] / 2;
    const int G = in_sizes[4];
    const int NF = N * FEAT;

    // workspace layout
    float* x    = (float*)d_ws;
    float* m    = x  + NF;
    float* vagg = m  + NF;
    float* t1   = vagg + NF;
    float* sbuf = t1 + NF;          // G*64
    float* cnt  = sbuf + G * FEAT;  // G   (contiguous with sbuf)
    float* solv = cnt + G;          // G*32
    _Float16* pb = (_Float16*)(solv + G * 32);  // 4 * 4096 halves

    const int TB = 256;
    auto blocks = [](int n, int b) { return (n + b - 1) / b; };

    // 1) pack g-network weights (f32 row-major -> f16 WMMA A layout, transposed
    //    GEMM). Layer-0 bias is folded into the padded K=63 row.
    for (int it = 0; it < 2; ++it) {
        const float* g0w = (const float*)d_in[11 + it * 10 + 2];
        const float* g0b = (const float*)d_in[11 + it * 10 + 3];
        const float* g1w = (const float*)d_in[11 + it * 10 + 4];
        const float* g1b = (const float*)d_in[11 + it * 10 + 5];
        pack_a_kernel<<<blocks(4096, TB), TB, 0, stream>>>(
            g0w, g0b, pb + (it * 2 + 0) * 4096, NRBF, 1);
        pack_a_kernel<<<blocks(4096, TB), TB, 0, stream>>>(
            g1w, g1b, pb + (it * 2 + 1) * 4096, FEAT, 0);
    }

    // 2) x = emb_z[z]
    embed_kernel<<<blocks(NF, TB), TB, 0, stream>>>(z, emb_z, x, N);

    // 3) interaction blocks
    for (int it = 0; it < 2; ++it) {
        const int b = 11 + it * 10;
        const float* lin1w = (const float*)d_in[b + 0];
        const float* lin1b = (const float*)d_in[b + 1];
        const float* g1b   = (const float*)d_in[b + 5];
        const float* mlp0w = (const float*)d_in[b + 6];
        const float* mlp0b = (const float*)d_in[b + 7];
        const float* mlp1w = (const float*)d_in[b + 8];
        const float* mlp1b = (const float*)d_in[b + 9];

        linear_kernel<<<blocks(NF, TB), TB, 0, stream>>>(x, lin1w, lin1b, m, nullptr,
                                                         N, FEAT, FEAT, 0);
        fill_kernel<<<blocks(NF, TB), TB, 0, stream>>>(vagg, NF, 0.0f);
        edge_kernel<<<512, EDGE_WAVES * 32, 0, stream>>>(
            ei, pos, m, pb + (it * 2 + 0) * 4096, pb + (it * 2 + 1) * 4096, g1b,
            vagg, E);
        linear_kernel<<<blocks(NF, TB), TB, 0, stream>>>(vagg, mlp0w, mlp0b, t1, nullptr,
                                                         N, FEAT, FEAT, 1);
        linear_kernel<<<blocks(NF, TB), TB, 0, stream>>>(t1, mlp1w, mlp1b, x, x,
                                                         N, FEAT, FEAT, 0);  // x += mlp(v)
    }

    // 4) post MLP: x -> t1 (ssp) -> m (reused as x2)
    linear_kernel<<<blocks(NF, TB), TB, 0, stream>>>(x, (const float*)d_in[31],
                                                     (const float*)d_in[32], t1, nullptr,
                                                     N, FEAT, FEAT, 1);
    linear_kernel<<<blocks(NF, TB), TB, 0, stream>>>(t1, (const float*)d_in[33],
                                                     (const float*)d_in[34], m, nullptr,
                                                     N, FEAT, FEAT, 0);

    // 5) scatter-mean pooling
    fill_kernel<<<blocks(G * FEAT + G, TB), TB, 0, stream>>>(sbuf, G * FEAT + G, 0.0f);
    scatter_kernel<<<blocks(NF, TB), TB, 0, stream>>>(m, batch, sbuf, cnt, N);

    // 6) solvent branch + head
    solvent_kernel<<<G, 64, 0, stream>>>(solvent, emb_s, sw1, sb1, sw2, sb2, solv);
    head_kernel<<<G, 128, 0, stream>>>(sbuf, cnt, solv,
                                       (const float*)d_in[35], (const float*)d_in[36],
                                       (const float*)d_in[37], (const float*)d_in[38],
                                       (const float*)d_in[39], (const float*)d_in[40],
                                       (float*)d_out);
}
